// HungarianMatcher_11647951307281
// MI455X (gfx1250) — matrix operand order repaired and checked by reading, not compiled, not run on hardware
//
#include <hip/hip_runtime.h>
#include <hip/hip_bf16.h>
#include <math.h>

// ---------------------------------------------------------------------------
// DETR Hungarian-matcher cost matrix on MI455X (gfx1250).
//   out[q, t] = 5*L1(norm boxes) - 2*softmax(logits)[q, label_t] - 2*GIoU(abs boxes)
//   q in [0,16000), t in [0,4096)  ->  262 MB f32 output (HBM-stream bound)
//
// Design:
//  * prep_pred: per-query softmax -> (hi,lo) f16 split tables (padded 80->96),
//               normalized pred box + area.
//  * prep_tgt : per-target normalized box + area + label.
//  * cost_main: 16x16 (q x t) WMMA tiles. cost_class = probs . onehot(label)
//               done on the matrix pipe via v_wmma_f32_16x16x32_f16 (hi + lo
//               accumulated into one f32 acc => ~f32 accuracy). GIoU/L1 done
//               on VALU per-lane in the native D-fragment layout, fused store.
// ---------------------------------------------------------------------------

typedef __attribute__((ext_vector_type(16))) _Float16 v16h;
typedef __attribute__((ext_vector_type(8)))  _Float16 v8h;
typedef __attribute__((ext_vector_type(8)))  float    v8f;

#define BQ_N   16000   // B*Q
#define BT_N   4096    // B*T
#define NCLS   80
#define NCP    96      // padded K (3 x 32)
#define QTILES 1000    // BQ_N / 16
#define TTILES 256     // BT_N / 16

// ---------------- prep: per-query softmax (hi/lo f16) + box precompute ------
__global__ __launch_bounds__(256) void prep_pred(
    const float* __restrict__ logits, const float* __restrict__ boxes,
    const float* __restrict__ imsz, _Float16* __restrict__ probHi,
    _Float16* __restrict__ probLo, float* __restrict__ predPre)
{
  int r = blockIdx.x * 256 + threadIdx.x;
  if (r >= BQ_N) return;
  const float* row = logits + (size_t)r * NCLS;

  float m = -INFINITY;
  #pragma unroll
  for (int i = 0; i < NCLS / 4; ++i) {
    float4 v4 = *(const float4*)(row + i * 4);
    m = fmaxf(m, fmaxf(fmaxf(v4.x, v4.y), fmaxf(v4.z, v4.w)));
  }
  float s = 0.f;
  #pragma unroll
  for (int i = 0; i < NCLS; ++i) s += expf(row[i] - m);
  float rs = 1.f / s;

  _Float16* ph = probHi + (size_t)r * NCP;
  _Float16* pl = probLo + (size_t)r * NCP;
  #pragma unroll
  for (int i = 0; i < NCLS; ++i) {
    float p = expf(row[i] - m) * rs;
    _Float16 h = (_Float16)p;
    ph[i] = h;
    pl[i] = (_Float16)(p - (float)h);   // residual -> ~f32 accuracy after hi+lo
  }
  #pragma unroll
  for (int i = NCLS; i < NCP; ++i) { ph[i] = (_Float16)0.f; pl[i] = (_Float16)0.f; }

  int b = r / 500;
  float4 bx = *(const float4*)(boxes + (size_t)r * 4);
  float4 sz = *(const float4*)(imsz + (size_t)b * 4);
  float* pp = predPre + (size_t)r * 8;
  pp[0] = bx.x / sz.x; pp[1] = bx.y / sz.y;
  pp[2] = bx.z / sz.z; pp[3] = bx.w / sz.w;
  pp[4] = (bx.z - bx.x) * (bx.w - bx.y);   // abs area
  pp[5] = 0.f; pp[6] = 0.f; pp[7] = 0.f;
}

// ---------------- prep: per-target box precompute ---------------------------
__global__ __launch_bounds__(256) void prep_tgt(
    const int* __restrict__ labels, const float* __restrict__ boxes,
    const float* __restrict__ imszt, float* __restrict__ tgtPre)
{
  int t = blockIdx.x * 256 + threadIdx.x;
  if (t >= BT_N) return;
  float4 bx = *(const float4*)(boxes + (size_t)t * 4);
  float4 sz = *(const float4*)(imszt + (size_t)t * 4);
  float* tp = tgtPre + (size_t)t * 8;
  tp[0] = bx.x / sz.x; tp[1] = bx.y / sz.y;
  tp[2] = bx.z / sz.z; tp[3] = bx.w / sz.w;
  tp[4] = (bx.z - bx.x) * (bx.w - bx.y);   // abs area
  tp[5] = __int_as_float(labels[t]);
  tp[6] = 0.f; tp[7] = 0.f;
}

// ---------------- main: fused WMMA class-cost + VALU L1/GIoU ---------------
__global__ __launch_bounds__(256) void cost_main(
    const float* __restrict__ pred_boxes, const float* __restrict__ tgt_boxes,
    const _Float16* __restrict__ probHi, const _Float16* __restrict__ probLo,
    const float* __restrict__ predPre, const float* __restrict__ tgtPre,
    float* __restrict__ out)
{
  const int lane  = threadIdx.x & 31;
  const int wave  = threadIdx.x >> 5;
  const int n     = lane & 15;       // column-in-tile (N)
  const int hi16  = lane >> 4;       // lane half
  const int tbase = blockIdx.x * 16; // this block's 16 targets
  const int tcol  = tbase + n;

  // per-lane target data (16 distinct targets per block, broadcast via cache)
  const float4 tb    = *(const float4*)(tgt_boxes + (size_t)tcol * 4);
  const float4 tn    = *(const float4*)(tgtPre   + (size_t)tcol * 8);
  const float  tarea = tgtPre[(size_t)tcol * 8 + 4];
  const int    tlab  = __float_as_int(tgtPre[(size_t)tcol * 8 + 5]);

  // one-hot B fragments (dense 16-bit B layout: lane holds N=lane%16,
  // K = 8*(lane/16) + e for e<8, 16 + 8*(lane/16) + (e-8) for e>=8), built once, reused for all q-tiles
  v16h bfrag[3];
  #pragma unroll
  for (int s = 0; s < 3; ++s) {
    #pragma unroll
    for (int e = 0; e < 16; ++e) {
      int k = s * 32 + hi16 * 8 + (e < 8 ? e : e + 8);
      bfrag[s][e] = (k == tlab) ? (_Float16)1.0f : (_Float16)0.0f;
    }
  }

  for (int qt = wave; qt < QTILES; qt += 8) {
    const int qbase = qt * 16;
    const _Float16* ph = probHi + (size_t)(qbase + n) * NCP;
    const _Float16* pl = probLo + (size_t)(qbase + n) * NCP;

    if (qt + 8 < QTILES) {  // next q-tile for this wave -> global_prefetch_b8
      __builtin_prefetch(probHi + (size_t)(qbase + 128 + n) * NCP, 0, 0);
      __builtin_prefetch(probLo + (size_t)(qbase + 128 + n) * NCP, 0, 0);
    }

    v8f acc = {};
    #pragma unroll
    for (int s = 0; s < 3; ++s) {
      // A layout (16-bit A 16x32): lanes 0-15 K{0..7,16..23}, lanes 16-31 K{8..15,24..31}
      const int off = s * 32 + hi16 * 8;
      v8h h0 = *(const v8h*)(ph + off);
      v8h h1 = *(const v8h*)(ph + off + 16);
      v8h l0 = *(const v8h*)(pl + off);
      v8h l1 = *(const v8h*)(pl + off + 16);
      v16h ahi = __builtin_shufflevector(h0, h1, 0,1,2,3,4,5,6,7,8,9,10,11,12,13,14,15);
      v16h alo = __builtin_shufflevector(l0, l1, 0,1,2,3,4,5,6,7,8,9,10,11,12,13,14,15);
      acc = __builtin_amdgcn_wmma_f32_16x16x32_f16(false, ahi, false, bfrag[s],
                                                   (short)0, acc, false, false);
      acc = __builtin_amdgcn_wmma_f32_16x16x32_f16(false, alo, false, bfrag[s],
                                                   (short)0, acc, false, false);
    }
    // acc[v] = softmax prob at this lane's (M = v + 8*hi16, N = n)

    #pragma unroll
    for (int v = 0; v < 8; ++v) {
      const int q = qbase + v + hi16 * 8;
      float4 pb = *(const float4*)(pred_boxes + (size_t)q * 4);
      float4 pn = *(const float4*)(predPre   + (size_t)q * 8);
      float parea = predPre[(size_t)q * 8 + 4];

      // L1 of normalized boxes
      float l1 = fabsf(pn.x - tn.x) + fabsf(pn.y - tn.y) +
                 fabsf(pn.z - tn.z) + fabsf(pn.w - tn.w);

      // GIoU of absolute boxes
      float iw    = fmaxf(fminf(pb.z, tb.z) - fmaxf(pb.x, tb.x), 0.f);
      float ih    = fmaxf(fminf(pb.w, tb.w) - fmaxf(pb.y, tb.y), 0.f);
      float inter = iw * ih;
      float uni   = parea + tarea - inter;
      float iou   = inter / uni;
      float ew    = fmaxf(fmaxf(pb.z, tb.z) - fminf(pb.x, tb.x), 0.f);
      float eh    = fmaxf(fmaxf(pb.w, tb.w) - fminf(pb.y, tb.y), 0.f);
      float ae    = ew * eh;
      float giou  = iou - (ae - uni) / ae;

      out[(size_t)q * BT_N + tcol] = 5.f * l1 - 2.f * acc[v] - 2.f * giou;
    }
  }
}

// ---------------------------------------------------------------------------
extern "C" void kernel_launch(void* const* d_in, const int* in_sizes, int n_in,
                              void* d_out, int out_size, void* d_ws, size_t ws_size,
                              hipStream_t stream) {
  const float* pred_logits = (const float*)d_in[0];  // [32,500,80]
  const float* pred_boxes  = (const float*)d_in[1];  // [32,500,4]
  const int*   tgt_labels  = (const int*)d_in[2];    // [32,128]
  const float* tgt_boxes   = (const float*)d_in[3];  // [32,128,4]
  const float* imsz        = (const float*)d_in[4];  // [32,4]
  const float* imszt       = (const float*)d_in[5];  // [32,128,4]
  float* out = (float*)d_out;                        // [32,500,4096]

  // workspace layout (~6.8 MB, 16B-aligned sections)
  char* ws = (char*)d_ws;
  _Float16* probHi  = (_Float16*)(ws);                 // 16000*96*2 = 3,072,000
  _Float16* probLo  = (_Float16*)(ws + 3072000);       // 3,072,000
  float*    predPre = (float*)(ws + 6144000);          // 16000*8*4 =   512,000
  float*    tgtPre  = (float*)(ws + 6656000);          //  4096*8*4 =   131,072

  prep_pred<<<(BQ_N + 255) / 256, 256, 0, stream>>>(
      pred_logits, pred_boxes, imsz, probHi, probLo, predPre);
  prep_tgt<<<(BT_N + 255) / 256, 256, 0, stream>>>(
      tgt_labels, tgt_boxes, imszt, tgtPre);
  cost_main<<<TTILES, 256, 0, stream>>>(
      pred_boxes, tgt_boxes, probHi, probLo, predPre, tgtPre, out);
}
